// Attention_57724360458278
// MI455X (gfx1250) — matrix operand order repaired and checked by reading, not compiled, not run on hardware
//
#include <hip/hip_runtime.h>

#define Bq  4
#define Tt  2048
#define Dd  512
#define Hh  8
#define DHd 64

typedef __attribute__((ext_vector_type(16))) _Float16 v16h;
typedef __attribute__((ext_vector_type(8)))  _Float16 v8h;
typedef __attribute__((ext_vector_type(8)))  float    v8f;

union V16 { v16h v; v8h h[2]; };

#if __has_builtin(__builtin_amdgcn_global_load_async_to_lds_b128)
#define HAVE_ASYNC_LDS 1
typedef int i4v __attribute__((vector_size(16)));
typedef __attribute__((address_space(1))) i4v* gv4;   // global int4*
typedef __attribute__((address_space(3))) i4v* lv4;   // LDS int4*
#endif

__device__ __forceinline__ v8f wmma16(v16h a, v16h b, v8f c) {
  // D = A(16x32 f16) * B(32x16 f16) + C(16x16 f32)
  return __builtin_amdgcn_wmma_f32_16x16x32_f16(false, a, false, b, (short)0, c,
                                                false, false);
}

// ---------------------------------------------------------------------------
// Kernel 1: fused QKV projection.  y = x @ W^T  with W (N,K) row-major.
// Output written as f16 in (b, h, t, dh) head layout, ready for WMMA frags.
// grid = (D/64 n-tiles, B*T/128 m-tiles, 3 weights), block = 256 (8 waves).
// ---------------------------------------------------------------------------
__global__ __launch_bounds__(256) void qkv_kernel(
    const float* __restrict__ x,  const float* __restrict__ Wq,
    const float* __restrict__ Wk, const float* __restrict__ Wv,
    _Float16* __restrict__ Qh, _Float16* __restrict__ Kh,
    _Float16* __restrict__ Vh)
{
  __shared__ _Float16 As[128 * 40];   // x tile  128(M) x 32(K), stride 40
  __shared__ _Float16 Bs[64 * 40];    // W tile   64(N) x 32(K), stride 40

  const int tid  = threadIdx.x;
  const int lane = tid & 31;
  const int wave = tid >> 5;
  const int mBase = blockIdx.y * 128;
  const int nBase = blockIdx.x * 64;
  const float*  W   = (blockIdx.z == 0) ? Wq : (blockIdx.z == 1) ? Wk : Wv;
  _Float16*     dst = (blockIdx.z == 0) ? Qh : (blockIdx.z == 1) ? Kh : Vh;

  const int m15 = lane & 15;
  const int hl  = lane >> 4;
  const int aRow = wave * 16 + m15;   // this wave owns rows [16w, 16w+16)
  const int aKb  = hl * 8;            // A-frag K chunks: base, base+16
  const int bKb  = hl * 8;            // B-frag K chunks: base, base+16

  v8f acc[4] = {};                    // 16 x 64 output strip per wave

  const int xRow = tid & 127, xCb = (tid >> 7) * 16;
  const int wRow = tid & 63,  wCb = (tid >> 6) * 8;

  for (int kt = 0; kt < Dd; kt += 32) {
    __syncthreads();
    {   // stage x tile (fp32 -> f16)
      const float* xp = x + (size_t)(mBase + xRow) * Dd + kt + xCb;
      for (int e = 0; e < 16; ++e)
        As[xRow * 40 + xCb + e] = (_Float16)xp[e];
    }
    {   // stage W tile: Bs[n][k] = W[nBase+n][kt+k]  (contiguous copy)
      const float* wp = W + (size_t)(nBase + wRow) * Dd + kt + wCb;
      for (int e = 0; e < 8; ++e)
        Bs[wRow * 40 + wCb + e] = (_Float16)wp[e];
    }
    __syncthreads();

    V16 a;
    a.h[0] = *(const v8h*)&As[aRow * 40 + aKb];
    a.h[1] = *(const v8h*)&As[aRow * 40 + aKb + 16];
    for (int n0 = 0; n0 < 4; ++n0) {
      const _Float16* bp = &Bs[(n0 * 16 + m15) * 40 + bKb];
      V16 bf;
      bf.h[0] = *(const v8h*)bp;
      bf.h[1] = *(const v8h*)(bp + 16);
      acc[n0] = wmma16(a.v, bf.v, acc[n0]);
    }
  }

  // C layout: vgpr r -> row r + 8*hl, col = lane&15
  for (int n0 = 0; n0 < 4; ++n0)
    for (int r = 0; r < 8; ++r) {
      int gm  = mBase + wave * 16 + r + 8 * hl;  // flat (b*T + t)
      int nch = nBase + n0 * 16 + m15;           // flat (h*64 + dh)
      int bb = gm >> 11, t = gm & (Tt - 1);
      int hh = nch >> 6, dh = nch & 63;
      dst[(((size_t)bb * Hh + hh) * Tt + t) * DHd + dh] = (_Float16)acc[n0][r];
    }
}

// ---------------------------------------------------------------------------
// Kernel 2: causal flash attention, one (b,h) and 128 query rows per block.
// 8 waves x 16 query rows; K/V tiles of 64 keys streamed through LDS.
// K tile uses the gfx1250 async global->LDS path (ASYNCcnt) when available.
// grid = (T/128, B*H), block = 256.
// ---------------------------------------------------------------------------
__global__ __launch_bounds__(256) void attn_kernel(
    const _Float16* __restrict__ Qh, const _Float16* __restrict__ Kh,
    const _Float16* __restrict__ Vh, float* __restrict__ out)
{
  __shared__ _Float16 Ks[64 * 72];       // [key][dh]      stride 72
  __shared__ _Float16 Vt[64 * 72];       // [dh][key]      stride 72 (V^T)
  __shared__ _Float16 Ps[8 * 16 * 72];   // per-wave P staging [16][64]

  const int tid  = threadIdx.x;
  const int lane = tid & 31;
  const int wave = tid >> 5;
  const int b = blockIdx.y >> 3, h = blockIdx.y & 7;
  const int qTile = blockIdx.x * 128;
  const int qbase = qTile + wave * 16;
  const size_t headBase = ((size_t)b * Hh + h) * Tt * DHd;

  const int m15 = lane & 15;
  const int hl  = lane >> 4;
  const int aKb = hl * 8;

  // Q fragments for this wave's 16 rows: two 16x32 A-frags (dh 0..31, 32..63)
  V16 qa[2];
  {
    const _Float16* qp = Qh + headBase + (size_t)(qbase + m15) * DHd;
    for (int t = 0; t < 2; ++t) {
      qa[t].h[0] = *(const v8h*)(qp + 32 * t + aKb);
      qa[t].h[1] = *(const v8h*)(qp + 32 * t + aKb + 16);
    }
  }

  v8f o[4] = {};                 // 16 x 64 output accumulator (4 C tiles)
  float mrow[8], lrow[8];        // row stats: row r + 8*hl per lane
  for (int r = 0; r < 8; ++r) { mrow[r] = -3.0e38f; lrow[r] = 0.0f; }

  const int ldRow = tid & 63, ldCb = (tid >> 6) * 16;
  const int jEnd = qTile + 128;  // causal: keys beyond block's last q unneeded
  _Float16* Psw = &Ps[wave * 16 * 72];
  const float scale = 0.125f;    // 1/sqrt(64)

  for (int jt = 0; jt < jEnd; jt += 64) {
    __syncthreads();
    {   // cooperative K (copy) and V (transpose) staging
      const _Float16* kp = Kh + headBase + (size_t)(jt + ldRow) * DHd + ldCb;
#ifdef HAVE_ASYNC_LDS
      // async DMA: global -> LDS without a VGPR round-trip (ASYNCcnt)
      __builtin_amdgcn_global_load_async_to_lds_b128(
          (gv4)kp,       (lv4)&Ks[ldRow * 72 + ldCb],     0, 0);
      __builtin_amdgcn_global_load_async_to_lds_b128(
          (gv4)(kp + 8), (lv4)&Ks[ldRow * 72 + ldCb + 8], 0, 0);
#else
      *(v8h*)&Ks[ldRow * 72 + ldCb]     = *(const v8h*)kp;
      *(v8h*)&Ks[ldRow * 72 + ldCb + 8] = *(const v8h*)(kp + 8);
#endif
      const _Float16* vp = Vh + headBase + (size_t)(jt + ldRow) * DHd + ldCb;
      V16 vv;
      vv.h[0] = *(const v8h*)vp;
      vv.h[1] = *(const v8h*)(vp + 8);
      for (int e = 0; e < 16; ++e)
        Vt[(ldCb + e) * 72 + ldRow] = vv.h[e >> 3][e & 7];
      if (jt + 64 < jEnd)
        __builtin_prefetch(kp + 64 * DHd, 0, 1);   // global_prefetch next tile
    }
#ifdef HAVE_ASYNC_LDS
    // each wave drains its own async copies before signaling the barrier
#if __has_builtin(__builtin_amdgcn_s_wait_asynccnt)
    __builtin_amdgcn_s_wait_asynccnt(0);
#else
    asm volatile("s_wait_asynccnt 0x0" ::: "memory");
#endif
#endif
    __syncthreads();

    if (jt > qbase + 15) continue;   // wave-uniform causal skip

    // ---- S = Q K^T : 4 key-subtiles, 2 WMMAs each (dh split 32+32) ----
    v8f s[4];
    for (int n0 = 0; n0 < 4; ++n0) {
      const _Float16* kp = &Ks[(n0 * 16 + m15) * 72 + hl * 8];
      V16 b0, b1;
      b0.h[0] = *(const v8h*)kp;        b0.h[1] = *(const v8h*)(kp + 16);
      b1.h[0] = *(const v8h*)(kp + 32); b1.h[1] = *(const v8h*)(kp + 48);
      v8f z = {};
      z     = wmma16(qa[0].v, b0.v, z);
      s[n0] = wmma16(qa[1].v, b1.v, z);
    }

    // ---- scale + causal mask ----
    for (int n0 = 0; n0 < 4; ++n0) {
      int key = jt + n0 * 16 + m15;
      for (int r = 0; r < 8; ++r) {
        int q = qbase + r + 8 * hl;
        float v = s[n0][r] * scale;
        if (key > q) v = -1.0e30f;
        s[n0][r] = v;
      }
    }

    // ---- online softmax: row max over 4 subtiles + 16-lane halves ----
    float mnew[8];
    for (int r = 0; r < 8; ++r)
      mnew[r] = fmaxf(fmaxf(s[0][r], s[1][r]), fmaxf(s[2][r], s[3][r]));
    for (int off = 1; off < 16; off <<= 1)
      for (int r = 0; r < 8; ++r)
        mnew[r] = fmaxf(mnew[r], __shfl_xor(mnew[r], off, 32));
    for (int r = 0; r < 8; ++r)
      mnew[r] = fmaxf(mnew[r], mrow[r]);

    float alpha[8], psum[8];
    for (int r = 0; r < 8; ++r) {
      alpha[r] = __expf(mrow[r] - mnew[r]);
      mrow[r]  = mnew[r];
      lrow[r] *= alpha[r];
      psum[r]  = 0.0f;
    }
    for (int d0 = 0; d0 < 4; ++d0)
      for (int r = 0; r < 8; ++r)
        o[d0][r] *= alpha[r];

    // ---- P = exp(S - m), stage to LDS (C-layout -> A-layout transpose) ----
    for (int n0 = 0; n0 < 4; ++n0)
      for (int r = 0; r < 8; ++r) {
        float p = __expf(s[n0][r] - mnew[r]);
        psum[r] += p;
        Psw[(r + 8 * hl) * 72 + n0 * 16 + m15] = (_Float16)p;
      }
    for (int off = 1; off < 16; off <<= 1)
      for (int r = 0; r < 8; ++r)
        psum[r] += __shfl_xor(psum[r], off, 32);
    for (int r = 0; r < 8; ++r) lrow[r] += psum[r];

    // ---- O += P V : P as two 16x32 A-frags, V^T as B-frags ----
    V16 pa[2];
    for (int t = 0; t < 2; ++t) {
      const _Float16* pp = &Psw[m15 * 72 + 32 * t + aKb];
      pa[t].h[0] = *(const v8h*)pp;
      pa[t].h[1] = *(const v8h*)(pp + 16);
    }
    for (int d0 = 0; d0 < 4; ++d0)
      for (int t = 0; t < 2; ++t) {
        const _Float16* vpp = &Vt[(d0 * 16 + m15) * 72 + 32 * t + hl * 8];
        V16 vb;
        vb.h[0] = *(const v8h*)vpp;
        vb.h[1] = *(const v8h*)(vpp + 16);
        o[d0] = wmma16(pa[t].v, vb.v, o[d0]);
      }
  }

  // ---- normalize and write (b, t, h*64+dh) fp32 ----
  for (int r = 0; r < 8; ++r) {
    float inv = 1.0f / lrow[r];
    int q = qbase + r + 8 * hl;
    float* op = out + ((size_t)b * Tt + q) * Dd + h * DHd;
    for (int d0 = 0; d0 < 4; ++d0)
      op[d0 * 16 + m15] = o[d0][r] * inv;
  }
}

// ---------------------------------------------------------------------------
extern "C" void kernel_launch(void* const* d_in, const int* in_sizes, int n_in,
                              void* d_out, int out_size, void* d_ws,
                              size_t ws_size, hipStream_t stream) {
  const float* x  = (const float*)d_in[0];
  const float* Wq = (const float*)d_in[1];
  const float* Wk = (const float*)d_in[2];
  const float* Wv = (const float*)d_in[3];
  float* out = (float*)d_out;

  const size_t headElems = (size_t)Bq * Hh * Tt * DHd;  // 4M f16 each
  _Float16* Qh = (_Float16*)d_ws;
  _Float16* Kh = Qh + headElems;
  _Float16* Vh = Kh + headElems;

  dim3 gA(Dd / 64, (Bq * Tt) / 128, 3);
  qkv_kernel<<<gA, 256, 0, stream>>>(x, Wq, Wk, Wv, Qh, Kh, Vh);

  dim3 gB(Tt / 128, Bq * Hh);
  attn_kernel<<<gB, 256, 0, stream>>>(Qh, Kh, Vh, out);
}
